// GNN_GCNLSTM_ea_fs_48653389529158
// MI455X (gfx1250) — matrix operand order repaired and checked by reading, not compiled, not run on hardware
//
#include <hip/hip_runtime.h>
#include <hip/hip_bf16.h>

#define N_NODES 50000
#define N_EDGES 1600000
#define F_NODE  32
#define F_EDGE  8
#define PERIODS 12
#define HIDDEN  64

typedef __attribute__((ext_vector_type(16))) __bf16 v16bf;
typedef __attribute__((ext_vector_type(8)))  __bf16 v8bf;
typedef __attribute__((ext_vector_type(8)))  float  v8f;

static __device__ __forceinline__ __bf16 f2bf(float x) {
  return (__bf16)x;                       // native v_cvt to bf16 on gfx1250
}
static __device__ __forceinline__ unsigned short f2bfbits(float x) {
  __bf16 h = (__bf16)x;
  return __builtin_bit_cast(unsigned short, h);
}
static __device__ __forceinline__ float fast_sigmoid(float x) {
  // 1 / (1 + e^-x) with HW v_exp_f32 + v_rcp_f32
  return __builtin_amdgcn_rcpf(1.0f + __expf(-x));
}
static __device__ __forceinline__ float fast_tanh(float x) {
  // (e^{2x} - 1) / (e^{2x} + 1), clamped so exp stays finite
  float xc = fminf(fmaxf(x, -15.0f), 15.0f);
  float t = __expf(2.0f * xc);
  return (t - 1.0f) * __builtin_amdgcn_rcpf(t + 1.0f);
}
static __device__ __forceinline__ void atomic_add_f32(float* p, float v) {
  unsafeAtomicAdd(p, v);   // -> global_atomic_add_f32
}

// ---------------- K0: deg = 1.0 (self loop) ----------------
__global__ __launch_bounds__(256) void k_initdeg(float* __restrict__ deg) {
  int n = blockIdx.x * 256 + threadIdx.x;
  if (n < N_NODES) deg[n] = 1.0f;
}

// ---------------- K1: ew = relu(ef@W_ef + b); deg[dst] += ew ----------------
__global__ __launch_bounds__(256) void k_edgew(const float* __restrict__ ef,
    const float* __restrict__ W_ef, const float* __restrict__ b_ef,
    const int* __restrict__ ei, float* __restrict__ ew, float* __restrict__ deg) {
  int e = blockIdx.x * 256 + threadIdx.x;
  if (e >= N_EDGES) return;
  float acc = b_ef[0];
#pragma unroll
  for (int j = 0; j < F_EDGE; ++j) acc += ef[(size_t)e * F_EDGE + j] * W_ef[j];
  acc = fmaxf(acc, 0.0f);
  ew[e] = acc;
  atomic_add_f32(&deg[ei[N_EDGES + e]], acc);
}

// ---------------- K2: dinv = rsqrt(deg) in place ----------------
__global__ __launch_bounds__(256) void k_rsqrt(float* __restrict__ deg) {
  int n = blockIdx.x * 256 + threadIdx.x;
  if (n < N_NODES) deg[n] = rsqrtf(deg[n]);
}

// ---------------- K3: xt[n][p][g] = sum_f x[n][f][p]*W_feat[f][g] + b[g] ------
__global__ __launch_bounds__(384) void k_feat(const float* __restrict__ x,
    const float* __restrict__ W_feat, const float* __restrict__ b_feat,
    float* __restrict__ xt) {
  __shared__ float Wf[F_NODE * F_NODE];
  __shared__ float bf[F_NODE];
  __shared__ float xs[F_NODE * PERIODS];
  int tid = threadIdx.x;
  for (int i = tid; i < F_NODE * F_NODE; i += 384) Wf[i] = W_feat[i];
  if (tid < F_NODE) bf[tid] = b_feat[tid];
  __syncthreads();
  int p = tid / F_NODE, g = tid % F_NODE;   // 384 = 12*32
  for (int n = blockIdx.x; n < N_NODES; n += gridDim.x) {
    xs[tid] = x[(size_t)n * 384 + tid];     // x is [n][f][p], p contiguous
    __syncthreads();
    float acc = bf[g];
#pragma unroll
    for (int f = 0; f < F_NODE; ++f) acc += xs[f * PERIODS + p] * Wf[f * F_NODE + g];
    xt[(size_t)n * 384 + p * F_NODE + g] = acc;   // [n][p][f]
    __syncthreads();
  }
}

// ---------------- Kw: fused gate weights, pre-swizzled to WMMA B-frag layout --
// Asw[g][t][lane][j]  = (Wg @ Lg_top)[K][col],  K=(j>>3)*16+(lane>>4)*8+(j&7), col=t*16+(lane&15)
// Bsw[g][kt][t][lane][j] = Lg_bot[K][col],      K=kt*32+(j>>3)*16+(lane>>4)*8+(j&7)
// csw[g][h] = bg @ Lg_top + Lg_b
__global__ __launch_bounds__(256) void k_weights(
    const float* Wz, const float* bz, const float* Wr, const float* br,
    const float* Wh, const float* bh,
    const float* LzW, const float* Lzb, const float* LrW, const float* Lrb,
    const float* LhW, const float* Lhb,
    unsigned short* __restrict__ Asw, unsigned short* __restrict__ Bsw,
    float* __restrict__ csw) {
  int idx = blockIdx.x * 256 + threadIdx.x;
  const float* Wg3[3] = {Wz, Wr, Wh};
  const float* Lg3[3] = {LzW, LrW, LhW};
  const float* bg3[3] = {bz, br, bh};
  const float* Lb3[3] = {Lzb, Lrb, Lhb};
  const int NA = 3 * 4 * 512, NB = 3 * 2 * 4 * 512;
  if (idx < NA) {
    int g = idx >> 11, rem = idx & 2047;
    int t = rem >> 9, q = rem & 511;
    int lane = q >> 4, j = q & 15;
    int K = ((j >> 3) << 4) + ((lane >> 4) << 3) + (j & 7);
    int col = t * 16 + (lane & 15);
    const float* Wg = Wg3[g]; const float* Lg = Lg3[g];
    float acc = 0.0f;
    for (int k = 0; k < HIDDEN; ++k) acc += Wg[K * HIDDEN + k] * Lg[k * HIDDEN + col];
    Asw[idx] = f2bfbits(acc);
  } else if (idx < NA + NB) {
    int o = idx - NA;
    int g = o / 4096, rem = o % 4096;
    int kt = rem >> 11, rem2 = rem & 2047;
    int t = rem2 >> 9, q = rem2 & 511;
    int lane = q >> 4, j = q & 15;
    int K = kt * 32 + ((j >> 3) << 4) + ((lane >> 4) << 3) + (j & 7);
    int col = t * 16 + (lane & 15);
    Bsw[o] = f2bfbits(Lg3[g][(HIDDEN + K) * HIDDEN + col]);
  } else if (idx < NA + NB + 192) {
    int o = idx - NA - NB;
    int g = o / HIDDEN, h = o % HIDDEN;
    float acc = Lb3[g][h];
    for (int k = 0; k < HIDDEN; ++k) acc += bg3[g][k] * Lg3[g][k * HIDDEN + h];
    csw[o] = acc;
  }
}

// ---------------- K4: U[p][n][f] = dinv[n]^2 * xt[n][p][f] ----------------
__global__ __launch_bounds__(256) void k_selfinit(const float* __restrict__ xt,
    const float* __restrict__ dinv, float* __restrict__ U) {
  size_t idx = (size_t)blockIdx.x * 256 + threadIdx.x;
  if (idx >= (size_t)N_NODES * 384) return;
  int n = (int)(idx / 384);
  int j = (int)(idx % 384);
  int p = j >> 5, f = j & 31;
  float dn = dinv[n];
  U[(size_t)p * ((size_t)N_NODES * 32) + (size_t)n * 32 + f] = dn * dn * xt[idx];
}

// ---------------- K5: edge scatter, one wave per edge ----------------
__global__ __launch_bounds__(256) void k_scatter(const int* __restrict__ ei,
    const float* __restrict__ ew, const float* __restrict__ dinv,
    const float* __restrict__ xt, float* __restrict__ U) {
  int wave = threadIdx.x >> 5, lane = threadIdx.x & 31;
  int e = blockIdx.x * 8 + wave;
  if (e >= N_EDGES) return;
  int s = ei[e], d = ei[N_EDGES + e];
  float w = dinv[s] * ew[e] * dinv[d];
  const float* xs = xt + (size_t)s * 384;
  float* ud = U + (size_t)d * 32 + lane;
#pragma unroll
  for (int p = 0; p < PERIODS; ++p) {
    float v = xs[p * 32 + lane] * w;                       // coalesced 128B
    atomic_add_f32(ud + (size_t)p * ((size_t)N_NODES * 32), v);  // contiguous 32 lanes
  }
}

// ---------------- K6: persistent WMMA GRU + output head ----------------
__global__ __launch_bounds__(128) void k_gru(
    const float* __restrict__ U, const unsigned short* __restrict__ Asw,
    const unsigned short* __restrict__ Bsw, const float* __restrict__ csw,
    const float* __restrict__ attention, const float* __restrict__ W_out,
    const float* __restrict__ b_out, float* __restrict__ out) {
  __shared__ alignas(32) unsigned short A_s[3 * 4 * 512];        // 12 KB
  __shared__ alignas(32) unsigned short B_s[3 * 2 * 4 * 512];    // 24 KB
  __shared__ float c_s[3 * HIDDEN];
  __shared__ alignas(32) float          H_s[4][16 * HIDDEN];     // 16 KB (f32 state)
  __shared__ alignas(32) unsigned short RH_s[4][16 * HIDDEN];    // 8 KB  (bf16 R*H)

  int tid = threadIdx.x;
  for (int i = tid; i < 3 * 4 * 512; i += 128) A_s[i] = Asw[i];
  for (int i = tid; i < 3 * 2 * 4 * 512; i += 128) B_s[i] = Bsw[i];
  for (int i = tid; i < 3 * HIDDEN; i += 128) c_s[i] = csw[i];

  // softmax(attention) redundantly per thread
  float att[PERIODS];
  float m = -1e30f;
#pragma unroll
  for (int p = 0; p < PERIODS; ++p) { att[p] = attention[p]; m = fmaxf(m, att[p]); }
  float ssum = 0.0f;
#pragma unroll
  for (int p = 0; p < PERIODS; ++p) { att[p] = __expf(att[p] - m); ssum += att[p]; }
  float rs = __builtin_amdgcn_rcpf(ssum);
#pragma unroll
  for (int p = 0; p < PERIODS; ++p) att[p] *= rs;

  int wave = tid >> 5, lane = tid & 31;
  int r = lane & 15, half = lane >> 4;
  int node0 = blockIdx.x * 64 + wave * 16;

  float* Hw = &H_s[wave][0];
  unsigned short* RHw = &RH_s[wave][0];
  for (int i = lane; i < 16 * HIDDEN; i += 32) Hw[i] = 0.0f;

  v8f hacc[4];
#pragma unroll
  for (int t = 0; t < 4; ++t)
#pragma unroll
    for (int i = 0; i < 8; ++i) hacc[t][i] = 0.0f;
  __syncthreads();

  int ndrow = node0 + r;
  if (ndrow > N_NODES - 1) ndrow = N_NODES - 1;
  const size_t plane = (size_t)N_NODES * 32;

  for (int p = 0; p < PERIODS; ++p) {
    // --- A-frag of u (16 nodes x 32K), bf16 from global (L2 resident) ---
    const float* Up = U + (size_t)p * plane + (size_t)ndrow * 32;
    v16bf ua;
#pragma unroll
    for (int g2 = 0; g2 < 2; ++g2) {
      float4 v0 = *(const float4*)(Up + half * 8 + g2 * 16);
      float4 v1 = *(const float4*)(Up + half * 8 + g2 * 16 + 4);
      ua[g2*8+0]=f2bf(v0.x); ua[g2*8+1]=f2bf(v0.y); ua[g2*8+2]=f2bf(v0.z); ua[g2*8+3]=f2bf(v0.w);
      ua[g2*8+4]=f2bf(v1.x); ua[g2*8+5]=f2bf(v1.y); ua[g2*8+6]=f2bf(v1.z); ua[g2*8+7]=f2bf(v1.w);
    }
    // --- A-frags of H (2 K-tiles of 32) ---
    v16bf ha[2];
#pragma unroll
    for (int kt = 0; kt < 2; ++kt)
#pragma unroll
      for (int g2 = 0; g2 < 2; ++g2) {
        const float* hp = Hw + r * HIDDEN + kt * 32 + half * 8 + g2 * 16;
        float4 v0 = *(const float4*)hp;
        float4 v1 = *(const float4*)(hp + 4);
        ha[kt][g2*8+0]=f2bf(v0.x); ha[kt][g2*8+1]=f2bf(v0.y); ha[kt][g2*8+2]=f2bf(v0.z); ha[kt][g2*8+3]=f2bf(v0.w);
        ha[kt][g2*8+4]=f2bf(v1.x); ha[kt][g2*8+5]=f2bf(v1.y); ha[kt][g2*8+6]=f2bf(v1.z); ha[kt][g2*8+7]=f2bf(v1.w);
      }

    v8f Z[4];
#pragma unroll
    for (int t = 0; t < 4; ++t) {
      int col = t * 16 + r;
      // Z gate
      v8f az;  float czb = c_s[0 * HIDDEN + col];
#pragma unroll
      for (int i = 0; i < 8; ++i) az[i] = czb;
      v16bf wb = *(const v16bf*)&A_s[((0 * 4 + t) * 32 + lane) * 16];
      az = __builtin_amdgcn_wmma_f32_16x16x32_bf16(false, ua, false, wb, (short)0, az, false, false);
      wb = *(const v16bf*)&B_s[(((0 * 2 + 0) * 4 + t) * 32 + lane) * 16];
      az = __builtin_amdgcn_wmma_f32_16x16x32_bf16(false, ha[0], false, wb, (short)0, az, false, false);
      wb = *(const v16bf*)&B_s[(((0 * 2 + 1) * 4 + t) * 32 + lane) * 16];
      az = __builtin_amdgcn_wmma_f32_16x16x32_bf16(false, ha[1], false, wb, (short)0, az, false, false);
#pragma unroll
      for (int i = 0; i < 8; ++i) Z[t][i] = fast_sigmoid(az[i]);
      // R gate -> RH staged as bf16 in LDS
      v8f ar;  float crb = c_s[1 * HIDDEN + col];
#pragma unroll
      for (int i = 0; i < 8; ++i) ar[i] = crb;
      wb = *(const v16bf*)&A_s[((1 * 4 + t) * 32 + lane) * 16];
      ar = __builtin_amdgcn_wmma_f32_16x16x32_bf16(false, ua, false, wb, (short)0, ar, false, false);
      wb = *(const v16bf*)&B_s[(((1 * 2 + 0) * 4 + t) * 32 + lane) * 16];
      ar = __builtin_amdgcn_wmma_f32_16x16x32_bf16(false, ha[0], false, wb, (short)0, ar, false, false);
      wb = *(const v16bf*)&B_s[(((1 * 2 + 1) * 4 + t) * 32 + lane) * 16];
      ar = __builtin_amdgcn_wmma_f32_16x16x32_bf16(false, ha[1], false, wb, (short)0, ar, false, false);
#pragma unroll
      for (int i = 0; i < 8; ++i) {
        float rr = fast_sigmoid(ar[i]);
        int rowD = half * 8 + i;                 // D-frag: row = half*8+i, col = lane&15 + 16t
        RHw[rowD * HIDDEN + col] = f2bfbits(rr * Hw[rowD * HIDDEN + col]);
      }
    }
    __syncthreads();

    // --- A-frags of R*H (bf16 straight from LDS) ---
    v16bf ra[2];
#pragma unroll
    for (int kt = 0; kt < 2; ++kt)
#pragma unroll
      for (int g2 = 0; g2 < 2; ++g2) {
        const v8bf ch = *(const v8bf*)(RHw + r * HIDDEN + kt * 32 + half * 8 + g2 * 16);
#pragma unroll
        for (int j = 0; j < 8; ++j) ra[kt][g2 * 8 + j] = ch[j];
      }

#pragma unroll
    for (int t = 0; t < 4; ++t) {
      int col = t * 16 + r;
      v8f ah;  float chb = c_s[2 * HIDDEN + col];
#pragma unroll
      for (int i = 0; i < 8; ++i) ah[i] = chb;
      v16bf wb = *(const v16bf*)&A_s[((2 * 4 + t) * 32 + lane) * 16];
      ah = __builtin_amdgcn_wmma_f32_16x16x32_bf16(false, ua, false, wb, (short)0, ah, false, false);
      wb = *(const v16bf*)&B_s[(((2 * 2 + 0) * 4 + t) * 32 + lane) * 16];
      ah = __builtin_amdgcn_wmma_f32_16x16x32_bf16(false, ra[0], false, wb, (short)0, ah, false, false);
      wb = *(const v16bf*)&B_s[(((2 * 2 + 1) * 4 + t) * 32 + lane) * 16];
      ah = __builtin_amdgcn_wmma_f32_16x16x32_bf16(false, ra[1], false, wb, (short)0, ah, false, false);
#pragma unroll
      for (int i = 0; i < 8; ++i) {
        float ht = fast_tanh(ah[i]);
        float z = Z[t][i];
        int rowD = half * 8 + i;
        float hold = Hw[rowD * HIDDEN + col];
        float hn = z * hold + (1.0f - z) * ht;
        Hw[rowD * HIDDEN + col] = hn;
        hacc[t][i] += att[p] * hn;
      }
    }
    __syncthreads();
  }

  // relu(Hacc) -> LDS, then output head: out[n][q] = h @ W_out + b_out
#pragma unroll
  for (int t = 0; t < 4; ++t)
#pragma unroll
    for (int i = 0; i < 8; ++i) {
      int rowD = half * 8 + i;
      Hw[rowD * HIDDEN + t * 16 + r] = fmaxf(hacc[t][i], 0.0f);
    }
  __syncthreads();
#pragma unroll
  for (int j = 0; j < 6; ++j) {
    int flat = lane * 6 + j;                 // 32*6 = 192 = 16 nodes * 12 outs
    int rr2 = flat / 12, q = flat - rr2 * 12;
    int node = node0 + rr2;
    if (node < N_NODES) {
      float acc = b_out[q];
#pragma unroll
      for (int k = 0; k < HIDDEN; ++k) acc += Hw[rr2 * HIDDEN + k] * W_out[k * PERIODS + q];
      out[(size_t)node * PERIODS + q] = acc;
    }
  }
}

extern "C" void kernel_launch(void* const* d_in, const int* in_sizes, int n_in,
                              void* d_out, int out_size, void* d_ws, size_t ws_size,
                              hipStream_t stream) {
  const float* x       = (const float*)d_in[0];
  const float* ef      = (const float*)d_in[1];
  const float* W_feat  = (const float*)d_in[2];
  const float* b_feat  = (const float*)d_in[3];
  const float* W_ef    = (const float*)d_in[4];
  const float* b_ef    = (const float*)d_in[5];
  const float* W_out   = (const float*)d_in[6];
  const float* b_out   = (const float*)d_in[7];
  const float* attn    = (const float*)d_in[8];
  const float* Wz      = (const float*)d_in[9];
  const float* bz      = (const float*)d_in[10];
  const float* Wr      = (const float*)d_in[11];
  const float* br      = (const float*)d_in[12];
  const float* Wh      = (const float*)d_in[13];
  const float* bh      = (const float*)d_in[14];
  const float* LzW     = (const float*)d_in[15];
  const float* Lzb     = (const float*)d_in[16];
  const float* LrW     = (const float*)d_in[17];
  const float* Lrb     = (const float*)d_in[18];
  const float* LhW     = (const float*)d_in[19];
  const float* Lhb     = (const float*)d_in[20];
  const int*   ei      = (const int*)d_in[21];
  float* out = (float*)d_out;

  char* ws = (char*)d_ws;
  size_t off = 0;
  auto alloc = [&](size_t bytes) {
    void* pp = ws + off;
    off = (off + bytes + 255) & ~(size_t)255;
    return pp;
  };
  float*          ew   = (float*)alloc((size_t)N_EDGES * 4);
  float*          deg  = (float*)alloc((size_t)N_NODES * 4);          // holds dinv after k_rsqrt
  float*          xt   = (float*)alloc((size_t)N_NODES * 384 * 4);    // [n][p][f]
  float*          U    = (float*)alloc((size_t)PERIODS * N_NODES * 32 * 4);
  unsigned short* Asw  = (unsigned short*)alloc((size_t)3 * 4 * 512 * 2);
  unsigned short* Bsw  = (unsigned short*)alloc((size_t)3 * 2 * 4 * 512 * 2);
  float*          csw  = (float*)alloc((size_t)3 * HIDDEN * 4);

  k_initdeg<<<(N_NODES + 255) / 256, 256, 0, stream>>>(deg);
  k_edgew<<<(N_EDGES + 255) / 256, 256, 0, stream>>>(ef, W_ef, b_ef, ei, ew, deg);
  k_rsqrt<<<(N_NODES + 255) / 256, 256, 0, stream>>>(deg);
  k_feat<<<2048, 384, 0, stream>>>(x, W_feat, b_feat, xt);
  k_weights<<<(3 * 4 * 512 + 3 * 2 * 4 * 512 + 192 + 255) / 256, 256, 0, stream>>>(
      Wz, bz, Wr, br, Wh, bh, LzW, Lzb, LrW, Lrb, LhW, Lhb, Asw, Bsw, csw);
  k_selfinit<<<(int)(((size_t)N_NODES * 384 + 255) / 256), 256, 0, stream>>>(xt, deg, U);
  k_scatter<<<(N_EDGES + 7) / 8, 256, 0, stream>>>(ei, ew, deg, xt, U);
  k_gru<<<(N_NODES + 63) / 64, 128, 0, stream>>>(U, Asw, Bsw, csw, attn, W_out, b_out, out);
}
